// Decoder_84267258348052
// MI455X (gfx1250) — compile-verified, matrix operand order checked by reading, not run on hardware
//
#include <hip/hip_runtime.h>
#include <hip/hip_bf16.h>
#include <math.h>

// ---------------- problem constants (match reference) ----------------
#define Bb     8
#define Ss     1024
#define Pp     1024
#define LOCAL  320
#define Hh     256
#define NHh    8
#define DHh    32
#define PFf    512
#define NLAY   2

typedef __attribute__((ext_vector_type(16))) _Float16 v16h;
typedef __attribute__((ext_vector_type(8)))  _Float16 v8h;
typedef __attribute__((ext_vector_type(4)))  _Float16 v4h;
typedef __attribute__((ext_vector_type(8)))  float    v8f;

#define WMMA16(a, b, c) \
  __builtin_amdgcn_wmma_f32_16x16x32_f16(false, (a), false, (b), (short)0, (c), false, false)

// CDNA5 async global->LDS copy (ASYNCcnt): each lane copies 16 contiguous
// bytes LDS[vdst] = MEM[vaddr].  Low 32 bits of a generic pointer into
// __shared__ equal the LDS byte offset (hardware truncates).
static __device__ inline void async_copy16(void* ldsDst, const void* gsrc) {
  asm volatile("global_load_async_to_lds_b128 %0, %1, off"
               :: "v"((unsigned)(unsigned long long)ldsDst), "v"(gsrc)
               : "memory");
}
static __device__ inline void async_wait0() {
  asm volatile("s_wait_asynccnt 0" ::: "memory");
}
static __device__ inline void async_wait3() {   // retire oldest stage, keep next in flight
  asm volatile("s_wait_asynccnt 3" ::: "memory");
}

// 16x32 f16 fragment, tile stored row-contiguous ([row][k], padded stride).
// CDNA5 A/B layout: lanes{0..15,16..31} hold rows 0..15; elem e<8 -> K=8*hf+e,
// e>=8 -> K=16+8*hf+(e-8).  Both 8-elem chunks contiguous -> ds_load_b128.
static __device__ inline v16h frag_ld(const _Float16* base, int stride, int row, int hf) {
  v16h f;
  const _Float16* p0 = base + row * stride + 8 * hf;
  const _Float16* p1 = p0 + 16;
#pragma unroll
  for (int e = 0; e < 8; ++e) { f[e] = p0[e]; f[8 + e] = p1[e]; }
  return f;
}
// Same fragment, tile stored [k][col] (strided gather along k).
static __device__ inline v16h frag_ld_strided(const _Float16* base, int stride, int col, int hf) {
  v16h f;
  const _Float16* p0 = base + (8 * hf) * stride + col;
  const _Float16* p1 = base + (16 + 8 * hf) * stride + col;
#pragma unroll
  for (int e = 0; e < 8; ++e) { f[e] = p0[e * stride]; f[8 + e] = p1[e * stride]; }
  return f;
}

#define AS_STR 40

// ---------------------------------------------------------------------
// f16 x f16 WMMA GEMM:  C = act(A @ Wt^T + bias)
// A:[M,K] lda f16 ; Wt:[N,K] ldw f16 (pre-transposed weights).
// 256 thr = 8 waves, block tile 64x128, wave tile 32x32.
// Both tiles staged with async-to-LDS, double-buffered (ASYNCcnt=3/stage).
// Out-of-range rows/cols clamp (garbage lands only in unsaved C elements).
// Outputs fp32 (Cf) and/or f16 (Ch).
// ---------------------------------------------------------------------
__global__ __launch_bounds__(256) void gemm_f16_kernel(
    const _Float16* __restrict__ A, int lda,
    const _Float16* __restrict__ Wt, int ldw,
    float* __restrict__ Cf, _Float16* __restrict__ Ch, int ldc,
    const float* __restrict__ bias,
    int M, int N, int K, int relu)
{
  __shared__ _Float16 As[2][64][AS_STR];
  __shared__ _Float16 Bs[2][128][AS_STR];

  const int t = threadIdx.x;
  const int rowBase = blockIdx.x * 64;
  const int colBase = blockIdx.y * 128;
  const int w  = t >> 5, lane = t & 31;
  const int wm = w >> 2, wn = w & 3;
  const int hf = lane >> 4, qn = lane & 15;

  const int ar = t >> 2, ac = (t & 3) * 8;     // A stage coords
  const int bn = t >> 1, bc = (t & 1) * 16;    // B stage coords
  const int gr = min(rowBase + ar, M - 1);     // clamp: ASYNCcnt constant
  const int gn = min(colBase + bn, N - 1);

  v8f acc00 = {0,0,0,0,0,0,0,0};
  v8f acc01 = {0,0,0,0,0,0,0,0};
  v8f acc10 = {0,0,0,0,0,0,0,0};
  v8f acc11 = {0,0,0,0,0,0,0,0};

  auto stage = [&](int buf, int k0) {
    async_copy16(&As[buf][ar][ac], A + (long)gr * lda + k0 + ac);
    const _Float16* wp = Wt + (long)gn * ldw + k0 + bc;
    async_copy16(&Bs[buf][bn][bc],     wp);
    async_copy16(&Bs[buf][bn][bc + 8], wp + 8);
  };

  stage(0, 0);
  int buf = 0;
  for (int k0 = 0; k0 < K; k0 += 32) {
    const bool more = (k0 + 32 < K);
    if (more) stage(buf ^ 1, k0 + 32);
    if (more) async_wait3(); else async_wait0();
    __syncthreads();

    v16h a0 = frag_ld(&As[buf][0][0], AS_STR, wm * 32 + qn,      hf);
    v16h a1 = frag_ld(&As[buf][0][0], AS_STR, wm * 32 + 16 + qn, hf);
    v16h b0 = frag_ld(&Bs[buf][0][0], AS_STR, wn * 32 + qn,      hf);
    v16h b1 = frag_ld(&Bs[buf][0][0], AS_STR, wn * 32 + 16 + qn, hf);

    acc00 = WMMA16(a0, b0, acc00);
    acc01 = WMMA16(a0, b1, acc01);
    acc10 = WMMA16(a1, b0, acc10);
    acc11 = WMMA16(a1, b1, acc11);
    __syncthreads();
    buf ^= 1;
  }

#pragma unroll
  for (int tm = 0; tm < 2; ++tm) {
#pragma unroll
    for (int tn = 0; tn < 2; ++tn) {
      v8f av = (tm == 0) ? (tn == 0 ? acc00 : acc01) : (tn == 0 ? acc10 : acc11);
      const int col = colBase + wn * 32 + tn * 16 + qn;
      if (col < N) {
        const float bv = bias ? bias[col] : 0.f;
#pragma unroll
        for (int r = 0; r < 8; ++r) {
          const int row = rowBase + wm * 32 + tm * 16 + 8 * hf + r;
          if (row < M) {
            float vv = av[r] + bv;
            if (relu) vv = fmaxf(vv, 0.f);
            if (Ch) Ch[(long)row * ldc + col] = (_Float16)vv;
            if (Cf) Cf[(long)row * ldc + col] = vv;
          }
        }
      }
    }
  }
}

// ---------------------------------------------------------------------
// Cross-attention energy: attn[b,h,q,k] = scale * Q . K   (f16 inputs)
// Single K-step (K = DH = 32); async-to-LDS staging.
// ---------------------------------------------------------------------
__global__ __launch_bounds__(256) void energy_f16_kernel(
    const _Float16* __restrict__ qh, const _Float16* __restrict__ kh,
    float* __restrict__ attn, float scale)
{
  __shared__ _Float16 As[64][AS_STR];    // Q tile [q][dh]
  __shared__ _Float16 Bs[128][AS_STR];   // K tile [key][dh]

  const int z = blockIdx.z;
  const int bz = z / NHh, hz = z % NHh;
  const long qbase = (long)bz * Ss * Hh + hz * DHh;
  const long kbase = (long)bz * Pp * Hh + hz * DHh;
  float* ab = attn + (long)z * Ss * Pp;

  const int t = threadIdx.x;
  const int rowBase = blockIdx.x * 64;
  const int colBase = blockIdx.y * 128;
  const int w  = t >> 5, lane = t & 31;
  const int wm = w >> 2, wn = w & 3;
  const int hf = lane >> 4, qn = lane & 15;

  { const int r = t >> 2, c = (t & 3) * 8;
    async_copy16(&As[r][c], qh + qbase + (long)(rowBase + r) * Hh + c); }
  { const int n = t >> 1, c0 = (t & 1) * 16;
    const _Float16* kp = kh + kbase + (long)(colBase + n) * Hh + c0;
    async_copy16(&Bs[n][c0], kp);
    async_copy16(&Bs[n][c0 + 8], kp + 8); }
  async_wait0();
  __syncthreads();

  const v8f z8 = {0,0,0,0,0,0,0,0};
  v16h a0 = frag_ld(&As[0][0], AS_STR, wm * 32 + qn,      hf);
  v16h a1 = frag_ld(&As[0][0], AS_STR, wm * 32 + 16 + qn, hf);
  v16h b0 = frag_ld(&Bs[0][0], AS_STR, wn * 32 + qn,      hf);
  v16h b1 = frag_ld(&Bs[0][0], AS_STR, wn * 32 + 16 + qn, hf);
  v8f c00 = WMMA16(a0, b0, z8);
  v8f c01 = WMMA16(a0, b1, z8);
  v8f c10 = WMMA16(a1, b0, z8);
  v8f c11 = WMMA16(a1, b1, z8);

#pragma unroll
  for (int tm = 0; tm < 2; ++tm) {
#pragma unroll
    for (int tn = 0; tn < 2; ++tn) {
      v8f av = (tm == 0) ? (tn == 0 ? c00 : c01) : (tn == 0 ? c10 : c11);
      const int col = colBase + wn * 32 + tn * 16 + qn;
#pragma unroll
      for (int r = 0; r < 8; ++r) {
        const int row = rowBase + wm * 32 + tm * 16 + 8 * hf + r;
        ab[(long)row * Pp + col] = av[r] * scale;
      }
    }
  }
}

// ---------------------------------------------------------------------
// PV GEMM: o[b,q,h*DH+d] = sum_k probs[b,h,q,k] * V[b,k,h*DH+d]
// probs fp32 gathered straight into the A fragment, V f16 async-to-LDS.
// 128 thr = 4 waves x 16 query rows; 32-key blocks.  f16 output.
// ---------------------------------------------------------------------
__global__ __launch_bounds__(128) void pv_f16_kernel(
    const float* __restrict__ probs, const _Float16* __restrict__ vh,
    _Float16* __restrict__ o)
{
  __shared__ _Float16 Vs[32][AS_STR];    // [key][dh]

  const int bh = blockIdx.y;
  const int bz = bh / NHh, hz = bh % NHh;
  const long pbase = (long)bh * Ss * Pp;
  const long vbase = (long)bz * Pp * Hh + hz * DHh;
  const long obase = (long)bz * Ss * Hh + hz * DHh;

  const int t = threadIdx.x;
  const int lane = t & 31, w = t >> 5;
  const int hf = lane >> 4, qn = lane & 15;
  const int qRow = blockIdx.x * 64 + w * 16;

  v8f acc0 = {0,0,0,0,0,0,0,0};
  v8f acc1 = {0,0,0,0,0,0,0,0};

  for (int kb = 0; kb < Pp; kb += 32) {
    { const int r = t >> 2, c = (t & 3) * 8;
      async_copy16(&Vs[r][c], vh + vbase + (long)(kb + r) * Hh + c); }

    v16h ap;
    { const float* p0 = probs + pbase + (long)(qRow + qn) * Pp + kb;
#pragma unroll
      for (int e = 0; e < 8; ++e) {
        ap[e]     = (_Float16)p0[8 * hf + e];
        ap[8 + e] = (_Float16)p0[16 + 8 * hf + e];
      }
    }
    async_wait0();
    __syncthreads();

    v16h bv0 = frag_ld_strided(&Vs[0][0], AS_STR, qn,      hf);
    v16h bv1 = frag_ld_strided(&Vs[0][0], AS_STR, 16 + qn, hf);
    acc0 = WMMA16(ap, bv0, acc0);
    acc1 = WMMA16(ap, bv1, acc1);
    __syncthreads();
  }

#pragma unroll
  for (int r = 0; r < 8; ++r) {
    const int row = qRow + 8 * hf + r;
    o[obase + (long)row * Hh + qn]      = (_Float16)acc0[r];
    o[obase + (long)row * Hh + 16 + qn] = (_Float16)acc1[r];
  }
}

// ---------------------------------------------------------------------
// Flash self-attention, f16 in / f16 out.  Async K/V staging.
// 128 thr = 4 waves x 16 query rows; 32-key blocks; online softmax.
// ---------------------------------------------------------------------
__global__ __launch_bounds__(128) void flash_attn_kernel(
    const _Float16* __restrict__ qh, const _Float16* __restrict__ kh,
    const _Float16* __restrict__ vh, _Float16* __restrict__ o, float scale)
{
  __shared__ _Float16 Kt[32][AS_STR];     // [key][dh]
  __shared__ _Float16 Vs[32][AS_STR];     // [key][dh]
  __shared__ _Float16 Pt[4][16][AS_STR];  // per-wave P [row][key]

  const int bh = blockIdx.y;
  const int bz = bh / NHh, hz = bh % NHh;
  const long base = (long)bz * Ss * Hh + hz * DHh;

  const int t = threadIdx.x;
  const int w = t >> 5, lane = t & 31;
  const int hf = lane >> 4, qn = lane & 15;
  const int qRow = blockIdx.x * 64 + w * 16;

  v16h aq;  // Q fragment, scale folded
  { const _Float16* qp = qh + base + (long)(qRow + qn) * Hh;
#pragma unroll
    for (int e = 0; e < 8; ++e) {
      aq[e]     = (_Float16)((float)qp[8 * hf + e]      * scale);
      aq[8 + e] = (_Float16)((float)qp[16 + 8 * hf + e] * scale);
    }
  }

  v8f acc0 = {0,0,0,0,0,0,0,0};
  v8f acc1 = {0,0,0,0,0,0,0,0};
  float mrun[8], lrun[8];
#pragma unroll
  for (int r = 0; r < 8; ++r) { mrun[r] = -1e30f; lrun[r] = 0.f; }

  for (int kb = 0; kb < Ss; kb += 32) {
    { const int r = t >> 2, c = (t & 3) * 8;
      const long go = base + (long)(kb + r) * Hh + c;
      async_copy16(&Kt[r][c], kh + go);
      async_copy16(&Vs[r][c], vh + go); }
    async_wait0();
    __syncthreads();

    v16h bk0 = frag_ld(&Kt[0][0], AS_STR, qn,      hf);
    v16h bk1 = frag_ld(&Kt[0][0], AS_STR, 16 + qn, hf);
    const v8f z8 = {0,0,0,0,0,0,0,0};
    v8f s0 = WMMA16(aq, bk0, z8);
    v8f s1 = WMMA16(aq, bk1, z8);

#pragma unroll
    for (int r = 0; r < 8; ++r) {
      float mx = fmaxf(s0[r], s1[r]);
#pragma unroll
      for (int m = 1; m < 16; m <<= 1) mx = fmaxf(mx, __shfl_xor(mx, m, 32));
      const float mnew = fmaxf(mrun[r], mx);
      const float corr = __expf(mrun[r] - mnew);
      const float p0 = __expf(s0[r] - mnew);
      const float p1 = __expf(s1[r] - mnew);
      float rs = p0 + p1;
#pragma unroll
      for (int m = 1; m < 16; m <<= 1) rs += __shfl_xor(rs, m, 32);
      lrun[r] = lrun[r] * corr + rs;
      mrun[r] = mnew;
      acc0[r] *= corr;
      acc1[r] *= corr;
      Pt[w][8 * hf + r][qn]      = (_Float16)p0;
      Pt[w][8 * hf + r][16 + qn] = (_Float16)p1;
    }

    v16h apf = frag_ld(&Pt[w][0][0], AS_STR, qn, hf);
    v16h bv0 = frag_ld_strided(&Vs[0][0], AS_STR, qn,      hf);
    v16h bv1 = frag_ld_strided(&Vs[0][0], AS_STR, 16 + qn, hf);
    acc0 = WMMA16(apf, bv0, acc0);
    acc1 = WMMA16(apf, bv1, acc1);
    __syncthreads();
  }

#pragma unroll
  for (int r = 0; r < 8; ++r) {
    const float inv = 1.0f / lrun[r];
    const int row = qRow + 8 * hf + r;
    o[base + (long)row * Hh + qn]      = (_Float16)(acc0[r] * inv);
    o[base + (long)row * Hh + 16 + qn] = (_Float16)(acc1[r] * inv);
  }
}

// ---------------- weight transpose+convert (one launch, 24 jobs) ------
struct TransJob  { const float* src; _Float16* dst; int K; int N; };
struct TransJobs { TransJob j[24]; };

__global__ __launch_bounds__(256) void transpose_w_kernel(TransJobs jobs)
{
  __shared__ float tile[32][33];
  const TransJob jb = jobs.j[blockIdx.z];
  const int n0 = blockIdx.x * 32, k0 = blockIdx.y * 32;
  if (n0 >= jb.N || k0 >= jb.K) return;
  const int tx = threadIdx.x & 31, ty = threadIdx.x >> 5;  // 32 x 8
#pragma unroll
  for (int i = 0; i < 32; i += 8) {
    const int k = k0 + ty + i, n = n0 + tx;
    tile[ty + i][tx] = (k < jb.K && n < jb.N) ? jb.src[(long)k * jb.N + n] : 0.f;
  }
  __syncthreads();
#pragma unroll
  for (int i = 0; i < 32; i += 8) {
    const int n = n0 + ty + i, k = k0 + tx;
    if (n < jb.N && k < jb.K) jb.dst[(long)n * jb.K + k] = (_Float16)tile[tx][ty + i];
  }
}

__global__ __launch_bounds__(256) void cvt_f32_f16_kernel(
    const float* __restrict__ in, _Float16* __restrict__ out, long n4)
{
  const long i = (long)blockIdx.x * 256 + threadIdx.x;
  if (i < n4) {
    const float4 v = ((const float4*)in)[i];
    v4h h; h[0] = (_Float16)v.x; h[1] = (_Float16)v.y;
           h[2] = (_Float16)v.z; h[3] = (_Float16)v.w;
    ((v4h*)out)[i] = h;
  }
}

// ------------------------- reduction kernels -------------------------
__global__ __launch_bounds__(256) void add_ln_kernel(
    float* __restrict__ x, _Float16* __restrict__ xh,
    const float* __restrict__ d,
    const float* __restrict__ g, const float* __restrict__ bt)
{
  __shared__ float red[256];
  const long row = blockIdx.x;
  const int t = threadIdx.x;
  const float v = x[row * Hh + t] + d[row * Hh + t];
  red[t] = v; __syncthreads();
  for (int s = 128; s > 0; s >>= 1) { if (t < s) red[t] += red[t + s]; __syncthreads(); }
  const float mu = red[0] / Hh; __syncthreads();
  const float c = v - mu;
  red[t] = c * c; __syncthreads();
  for (int s = 128; s > 0; s >>= 1) { if (t < s) red[t] += red[t + s]; __syncthreads(); }
  const float var = red[0] / Hh;
  const float o = c * rsqrtf(var + 1e-5f) * g[t] + bt[t];
  x[row * Hh + t]  = o;
  xh[row * Hh + t] = (_Float16)o;
}

// Fused single-pass softmax over rows of exactly 1024 (1 read + 1 write).
__global__ __launch_bounds__(256) void softmax1024_kernel(float* __restrict__ data)
{
  __shared__ float red[256];
  float* p = data + (long)blockIdx.x * 1024;
  const int t = threadIdx.x;
  float4 v = *(const float4*)(p + t * 4);
  float mx = fmaxf(fmaxf(v.x, v.y), fmaxf(v.z, v.w));
  red[t] = mx; __syncthreads();
  for (int s = 128; s > 0; s >>= 1) { if (t < s) red[t] = fmaxf(red[t], red[t + s]); __syncthreads(); }
  mx = red[0]; __syncthreads();
  v.x = __expf(v.x - mx); v.y = __expf(v.y - mx);
  v.z = __expf(v.z - mx); v.w = __expf(v.w - mx);
  red[t] = v.x + v.y + v.z + v.w; __syncthreads();
  for (int s = 128; s > 0; s >>= 1) { if (t < s) red[t] += red[t + s]; __syncthreads(); }
  const float inv = 1.0f / red[0];
  v.x *= inv; v.y *= inv; v.z *= inv; v.w *= inv;
  *(float4*)(p + t * 4) = v;
}

__global__ __launch_bounds__(256) void row_norm_kernel(const float* __restrict__ x,
                                                       float* __restrict__ out)
{
  __shared__ float red[256];
  const long row = blockIdx.x;
  const int t = threadIdx.x;
  const float v = x[row * Hh + t];
  red[t] = v * v; __syncthreads();
  for (int s = 128; s > 0; s >>= 1) { if (t < s) red[t] += red[t + s]; __syncthreads(); }
  if (t == 0) out[row] = sqrtf(red[0]);
}

__global__ __launch_bounds__(256) void pool_kernel(const float* __restrict__ x,
                                                   const float* __restrict__ wts,
                                                   float* __restrict__ outf,
                                                   _Float16* __restrict__ outh)
{
  const int bz = blockIdx.x, t = threadIdx.x;
  float acc = 0.f;
  for (int s = 0; s < Ss; ++s)
    acc += wts[bz * Ss + s] * x[((long)bz * Ss + s) * Hh + t];
  outf[bz * Hh + t] = acc;
  outh[bz * Hh + t] = (_Float16)acc;
}

// ---------------------------------------------------------------------
extern "C" void kernel_launch(void* const* d_in, const int* in_sizes, int n_in,
                              void* d_out, int out_size, void* d_ws, size_t ws_size,
                              hipStream_t stream)
{
  (void)in_sizes; (void)n_in; (void)out_size; (void)ws_size;

  const float* trg = (const float*)d_in[0];
  const float* src = (const float*)d_in[1];
  auto IN = [&](int i) { return (const float*)d_in[i]; };
  int ix = 2;
  const float* ft_w = IN(ix++); const float* ft_b = IN(ix++);
  struct AttnP { const float *wq,*bq,*wk,*bk,*wv,*bv,*wo,*bo; };
  AttnP sa[NLAY], ea[NLAY];
  const float *w1[NLAY], *b1[NLAY], *w2[NLAY], *b2[NLAY];
  const float *l1g[NLAY], *l1b[NLAY], *l2g[NLAY], *l2b[NLAY], *l3g[NLAY], *l3b[NLAY];
  for (int l = 0; l < NLAY; ++l) {
    sa[l] = {IN(ix), IN(ix+1), IN(ix+2), IN(ix+3), IN(ix+4), IN(ix+5), IN(ix+6), IN(ix+7)}; ix += 8;
    ea[l] = {IN(ix), IN(ix+1), IN(ix+2), IN(ix+3), IN(ix+4), IN(ix+5), IN(ix+6), IN(ix+7)}; ix += 8;
    w1[l] = IN(ix++); b1[l] = IN(ix++); w2[l] = IN(ix++); b2[l] = IN(ix++);
    l1g[l] = IN(ix++); l1b[l] = IN(ix++); l2g[l] = IN(ix++); l2b[l] = IN(ix++);
    l3g[l] = IN(ix++); l3b[l] = IN(ix++);
  }
  const float* fc1_w = IN(ix++); const float* fc1_b = IN(ix++);
  const float* fc2_w = IN(ix++); const float* fc2_b = IN(ix++);
  const float* fc3_w = IN(ix++); const float* fc3_b = IN(ix++);

  // ---- workspace: fp32 region then f16 region
  const long BSH = (long)Bb * Ss * Hh;
  float* wsf = (float*)d_ws;
  float* x   = wsf;                            // [B,S,H]
  float* t2  = x + BSH;                        // residual delta
  float* wts = t2 + BSH;                       // [B,S]
  _Float16* f16p = (_Float16*)(wts + (long)Bb * Ss);
  long fo = 0;
  auto halloc = [&](long n) { _Float16* p = f16p + fo; fo += n; return p; };
  _Float16* xh   = halloc(BSH);
  _Float16* qh   = halloc(BSH);
  _Float16* kh   = halloc(BSH);
  _Float16* vh   = halloc(BSH);
  _Float16* th   = halloc(BSH);                // attention head outputs
  _Float16* ffh  = halloc((long)Bb * Ss * PFf);
  _Float16* trgh = halloc((long)Bb * Ss * LOCAL);
  _Float16* srch = halloc((long)Bb * Pp * Hh);
  _Float16* sumh = halloc(Bb * Hh);
  _Float16* h1h  = halloc(Bb * 256);
  _Float16* h2h  = halloc(Bb * 64);

  // transposed f16 weights [N][K]
  TransJobs tj; int jn = 0;
  auto addT = [&](const float* s, int K, int N) {
    _Float16* d = halloc((long)K * N);
    tj.j[jn++] = TransJob{s, d, K, N};
    return d;
  };
  _Float16* ftT = addT(ft_w, LOCAL, Hh);
  struct AttnT { _Float16 *wqT,*wkT,*wvT,*woT; };
  AttnT saT[NLAY], eaT[NLAY];
  _Float16 *w1T[NLAY], *w2T[NLAY];
  for (int l = 0; l < NLAY; ++l) {
    saT[l] = {addT(sa[l].wq, Hh, Hh), addT(sa[l].wk, Hh, Hh),
              addT(sa[l].wv, Hh, Hh), addT(sa[l].wo, Hh, Hh)};
    eaT[l] = {addT(ea[l].wq, Hh, Hh), addT(ea[l].wk, Hh, Hh),
              addT(ea[l].wv, Hh, Hh), addT(ea[l].wo, Hh, Hh)};
    w1T[l] = addT(w1[l], Hh, PFf);
    w2T[l] = addT(w2[l], PFf, Hh);
  }
  _Float16* fc1T = addT(fc1_w, Hh, 256);
  _Float16* fc2T = addT(fc2_w, 256, 64);
  _Float16* fc3T = addT(fc3_w, 64, 2);

  // ---- output layout: sum_features | attn | logits
  float* outF   = (float*)d_out;
  float* sumf   = outF;
  float* attn   = outF + (long)Bb * Hh;
  float* logits = attn + (long)Bb * NHh * Ss * Pp;

  const float iscale = 0.17677669529663687f;   // 1/sqrt(DH)

  // one-time conversions
  transpose_w_kernel<<<dim3(16, 16, 24), 256, 0, stream>>>(tj);
  cvt_f32_f16_kernel<<<(Bb * Ss * LOCAL / 4 + 255) / 256, 256, 0, stream>>>(
      trg, trgh, (long)Bb * Ss * LOCAL / 4);
  cvt_f32_f16_kernel<<<(Bb * Pp * Hh / 4 + 255) / 256, 256, 0, stream>>>(
      src, srch, (long)Bb * Pp * Hh / 4);

  auto gemm = [&](const _Float16* A, int lda, const _Float16* Wt,
                  const float* bias, float* Cf, _Float16* Ch, int ldc,
                  int M, int N, int K, int relu) {
    dim3 g((M + 63) / 64, (N + 127) / 128, 1);
    gemm_f16_kernel<<<g, 256, 0, stream>>>(A, lda, Wt, K, Cf, Ch, ldc, bias,
                                           M, N, K, relu);
  };

  // feature transform: x = trg @ ft_w + ft_b   (fp32 master + f16 mirror)
  gemm(trgh, LOCAL, ftT, ft_b, x, xh, Hh, Bb * Ss, Hh, LOCAL, 0);

  for (int l = 0; l < NLAY; ++l) {
    // ---- self attention (flash) ----
    gemm(xh, Hh, saT[l].wqT, sa[l].bq, nullptr, qh, Hh, Bb * Ss, Hh, Hh, 0);
    gemm(xh, Hh, saT[l].wkT, sa[l].bk, nullptr, kh, Hh, Bb * Ss, Hh, Hh, 0);
    gemm(xh, Hh, saT[l].wvT, sa[l].bv, nullptr, vh, Hh, Bb * Ss, Hh, Hh, 0);
    flash_attn_kernel<<<dim3(Ss / 64, Bb * NHh), 128, 0, stream>>>(qh, kh, vh, th, iscale);
    gemm(th, Hh, saT[l].woT, sa[l].bo, t2, nullptr, Hh, Bb * Ss, Hh, Hh, 0);
    add_ln_kernel<<<Bb * Ss, 256, 0, stream>>>(x, xh, t2, l1g[l], l1b[l]);

    // ---- cross attention (materialized into d_out attn region) ----
    gemm(xh,   Hh, eaT[l].wqT, ea[l].bq, nullptr, qh, Hh, Bb * Ss, Hh, Hh, 0);
    gemm(srch, Hh, eaT[l].wkT, ea[l].bk, nullptr, kh, Hh, Bb * Pp, Hh, Hh, 0);
    gemm(srch, Hh, eaT[l].wvT, ea[l].bv, nullptr, vh, Hh, Bb * Pp, Hh, Hh, 0);
    energy_f16_kernel<<<dim3(Ss / 64, Pp / 128, Bb * NHh), 256, 0, stream>>>(
        qh, kh, attn, iscale);
    softmax1024_kernel<<<Bb * NHh * Ss, 256, 0, stream>>>(attn);
    pv_f16_kernel<<<dim3(Ss / 64, Bb * NHh), 128, 0, stream>>>(attn, vh, th);
    gemm(th, Hh, eaT[l].woT, ea[l].bo, t2, nullptr, Hh, Bb * Ss, Hh, Hh, 0);
    add_ln_kernel<<<Bb * Ss, 256, 0, stream>>>(x, xh, t2, l2g[l], l2b[l]);

    // ---- FFN ----
    gemm(xh,  Hh,  w1T[l], b1[l], nullptr, ffh, PFf, Bb * Ss, PFf, Hh, 1);
    gemm(ffh, PFf, w2T[l], b2[l], t2, nullptr, Hh, Bb * Ss, Hh, PFf, 0);
    add_ln_kernel<<<Bb * Ss, 256, 0, stream>>>(x, xh, t2, l3g[l], l3b[l]);
  }

  // ---- pooling: softmax over L2 norms, weighted sum ----
  row_norm_kernel<<<Bb * Ss, 256, 0, stream>>>(x, wts);
  softmax1024_kernel<<<Bb, 256, 0, stream>>>(wts);     // S == 1024
  pool_kernel<<<Bb, 256, 0, stream>>>(x, wts, sumf, sumh);

  // ---- MLP head ----
  gemm(sumh, Hh,  fc1T, fc1_b, nullptr, h1h, 256, Bb, 256, Hh, 1);
  gemm(h1h,  256, fc2T, fc2_b, nullptr, h2h, 64, Bb, 64, 256, 1);
  gemm(h2h,  64,  fc3T, fc3_b, logits, nullptr, 2, Bb, 2, 64, 0);
}
